// MixerBlock_12481174962963
// MI455X (gfx1250) — matrix operand order repaired and checked by reading, not compiled, not run on hardware
//
#include <hip/hip_runtime.h>
#include <cstdint>

// Problem sizes fixed by the reference
#define Bv   4
#define Ev   1024
#define Sv   2048
#define Hv   16
#define Mv   (Bv*Sv)          // 8192 GEMM rows
// GEMM tiling
#define BM   64
#define BN   256
#define BK   32
#define LPAD 40               // LDS row pitch (elements) to spread banks

typedef __attribute__((ext_vector_type(16))) __bf16 v16bf;
typedef __attribute__((ext_vector_type(8)))  __bf16 v8bf;
typedef __attribute__((ext_vector_type(8)))  float  v8f;
typedef __attribute__((ext_vector_type(4)))  int    v4i;

// ---- CDNA5 async global->LDS copy (ASYNCcnt path), with sync fallback ----
#if defined(__gfx1250__) && __has_builtin(__builtin_amdgcn_global_load_async_to_lds_b128)
#define USE_ASYNC 1
#else
#define USE_ASYNC 0
#endif

#if USE_ASYNC
typedef __attribute__((address_space(1))) v4i gv4i;   // global int4
typedef __attribute__((address_space(3))) v4i lv4i;   // LDS int4
#endif

__device__ __forceinline__ void copy16(const __bf16* g, __bf16* l) {
#if USE_ASYNC
  __builtin_amdgcn_global_load_async_to_lds_b128(
      (gv4i*)(uintptr_t)g, (lv4i*)(uintptr_t)l, 0, 0);
#else
  *(v8bf*)l = *(const v8bf*)g;
#endif
}

__device__ __forceinline__ void async_drain() {
#if USE_ASYNC
  asm volatile("s_wait_asynccnt 0x0" ::: "memory");
#endif
}

// ---------------- fp32 -> bf16 weight convert (layout preserved) ----------------
__global__ __launch_bounds__(256) void k_conv_w(const float* __restrict__ w,
                                                __bf16* __restrict__ o, int n) {
  int i = blockIdx.x * 256 + threadIdx.x;
  if (i < n) o[i] = (__bf16)w[i];
}

// ------- x (B,E,S) fp32  ->  A (B*S, E) bf16 row-major (tiled transpose) -------
__global__ __launch_bounds__(256) void k_conv_x(const float* __restrict__ x,
                                                __bf16* __restrict__ a) {
  __shared__ float tile[32][33];
  const int b  = blockIdx.z;
  const int e0 = blockIdx.y * 32;
  const int t0 = blockIdx.x * 32;
  for (int r = threadIdx.y; r < 32; r += 8)
    tile[r][threadIdx.x] = x[((size_t)b*Ev + e0 + r)*Sv + t0 + threadIdx.x];
  __syncthreads();
  for (int r = threadIdx.y; r < 32; r += 8)
    a[((size_t)(b*Sv + t0 + r))*Ev + e0 + threadIdx.x] = (__bf16)tile[threadIdx.x][r];
}

// ---------------- bf16 WMMA GEMM: C[m,n] = sum_k A[m,k] * W[n,k] + bias[n] ------
// A: (Mv x Ev) bf16 row-major.  W: (Ev x Ev) bf16 row-major, indexed [n][k].
// Double-buffered LDS; tile K+1 staged asynchronously while K's WMMAs execute.
// OUT_MODE 0: out[m*Ev + n] (fp32, row-major)
// OUT_MODE 1: out[(b*Ev + n)*Sv + t] with m = b*Sv + t  (final b e t transpose)
template <int OUT_MODE>
__global__ __launch_bounds__(256)
void k_gemm(const __bf16* __restrict__ A, const __bf16* __restrict__ W,
            const float* __restrict__ bias, float* __restrict__ out) {
  __shared__ __bf16 lA[2][BM][LPAD];
  __shared__ __bf16 lB[2][BN][LPAD];

  const int tid   = threadIdx.x;
  const int lane  = tid & 31;
  const int wave  = tid >> 5;       // 0..7
  const int waveM = wave >> 2;      // 0..1  (32 rows each)
  const int waveN = wave & 3;       // 0..3  (64 cols each)
  const int half  = lane >> 4;      // lane group (K-half select)
  const int mr    = lane & 15;      // row/col within fragment
  const int m0 = blockIdx.y * BM;
  const int n0 = blockIdx.x * BN;

  const int arow = tid >> 2;        // 0..63
  const int achk = (tid & 3) * 8;   // 0/8/16/24 (bf16 elements)

  v8f acc[2][4] = {};

  auto stage = [&](int buf, int k0) {
    // A tile (64x32): one 16B chunk per thread
    copy16(&A[(size_t)(m0 + arow)*Ev + k0 + achk], &lA[buf][arow][achk]);
    // B tile (256x32, n-major so fragments read contiguous K): one row per thread
    const __bf16* src = &W[(size_t)(n0 + tid)*Ev + k0];
    copy16(&src[0],  &lB[buf][tid][0]);
    copy16(&src[8],  &lB[buf][tid][8]);
    copy16(&src[16], &lB[buf][tid][16]);
    copy16(&src[24], &lB[buf][tid][24]);
  };

  stage(0, 0);
  async_drain();
  __syncthreads();

  int cur = 0;
  for (int k0 = 0; k0 < Ev; k0 += BK) {
    // Prefetch next K tile into the other buffer while we compute this one.
    if (k0 + BK < Ev) stage(cur ^ 1, k0 + BK);

    // A fragment per ISA layout: lanes 0-15 hold K {0..7,16..23}, lanes 16-31 K {8..15,24..31}
    v16bf af[2];
#pragma unroll
    for (int i = 0; i < 2; ++i) {
      const __bf16* ar = &lA[cur][waveM*32 + i*16 + mr][0];
      v8bf lo = *(const v8bf*)&ar[8*half];
      v8bf hi = *(const v8bf*)&ar[16 + 8*half];
      af[i] = __builtin_shufflevector(lo, hi, 0,1,2,3,4,5,6,7,8,9,10,11,12,13,14,15);
    }
    // B fragment: lanes 0-15 K {0..7,16..23}, lanes 16-31 K {8..15,24..31} (matches A)
    v16bf bfr[4];
#pragma unroll
    for (int j = 0; j < 4; ++j) {
      const __bf16* br = &lB[cur][waveN*64 + j*16 + mr][0];
      v8bf lo = *(const v8bf*)&br[8*half];
      v8bf hi = *(const v8bf*)&br[16 + 8*half];
      bfr[j] = __builtin_shufflevector(lo, hi, 0,1,2,3,4,5,6,7,8,9,10,11,12,13,14,15);
    }
#pragma unroll
    for (int i = 0; i < 2; ++i)
#pragma unroll
      for (int j = 0; j < 4; ++j)
        acc[i][j] = __builtin_amdgcn_wmma_f32_16x16x32_bf16(
            false, af[i], false, bfr[j], (short)0, acc[i][j], false, false);

    async_drain();        // next buffer's async copies complete (hidden behind WMMAs)
    __syncthreads();      // all waves' copies visible before the buffer swap
    cur ^= 1;
  }

  // Epilogue.  C layout: VGPR r holds (M = r + 8*half, N = mr) of each 16x16 tile.
#pragma unroll
  for (int i = 0; i < 2; ++i) {
#pragma unroll
    for (int j = 0; j < 4; ++j) {
      const int gn = n0 + waveN*64 + j*16 + mr;
      const float bs = bias[gn];
#pragma unroll
      for (int r = 0; r < 8; ++r) {
        const int gm = m0 + waveM*32 + i*16 + 8*half + r;
        const float v = acc[i][j][r] + bs;
        if (OUT_MODE == 0) {
          out[(size_t)gm*Ev + gn] = v;
        } else {
          const int bb = gm >> 11;          // / Sv
          const int t  = gm & (Sv - 1);
          out[((size_t)bb*Ev + gn)*Sv + t] = v;   // 8 contiguous floats along t per lane
        }
      }
    }
  }
}

// -------- causal head-mixing as chunked prefix sums (2-pass) --------
// p[b,h,c,t] = Y[(b*S+t)*E + h*64 + c]
// col heads (h<8):  out = mix_w[h,s] * cumsum_t(p)      + mix_b[h,s]
// row heads (h>=8): out = cumsum_t(mix_w[h,t] * p)      + mix_b[h,s]
#define NCHUNK 16
#define CLEN   (Sv / NCHUNK)   // 128

__global__ __launch_bounds__(64) void k_mix_partial(const float* __restrict__ Y,
                                                    const float* __restrict__ mix_w,
                                                    float* __restrict__ part) {
  const int bh = blockIdx.x, chunk = blockIdx.y, c = threadIdx.x;
  const int b = bh >> 4, h = bh & 15;
  const bool rowhead = h >= (Hv/2);
  const float* wv = mix_w + (size_t)h*Sv;
  const int i = h*64 + c;
  float s = 0.f;
  const int t0 = chunk * CLEN;
  for (int t = t0; t < t0 + CLEN; ++t) {
    float p = Y[(size_t)(b*Sv + t)*Ev + i];
    s += rowhead ? wv[t]*p : p;
  }
  part[((size_t)bh*NCHUNK + chunk)*64 + c] = s;
}

__global__ __launch_bounds__(64) void k_mix_final(const float* __restrict__ Y,
                                                  const float* __restrict__ mix_w,
                                                  const float* __restrict__ mix_b,
                                                  const float* __restrict__ part,
                                                  __bf16* __restrict__ outA) {
  const int bh = blockIdx.x, chunk = blockIdx.y, c = threadIdx.x;
  const int b = bh >> 4, h = bh & 15;
  const bool rowhead = h >= (Hv/2);
  const float* wv = mix_w + (size_t)h*Sv;
  const float* bv = mix_b + (size_t)h*Sv;
  const int i = h*64 + c;
  float acc = 0.f;
  for (int cc = 0; cc < chunk; ++cc)             // exclusive prefix of chunk sums
    acc += part[((size_t)bh*NCHUNK + cc)*64 + c];
  const int t0 = chunk * CLEN;
  for (int t = t0; t < t0 + CLEN; ++t) {
    float p = Y[(size_t)(b*Sv + t)*Ev + i];
    acc += rowhead ? wv[t]*p : p;                // inclusive running value
    float o = (rowhead ? acc : wv[t]*acc) + bv[t];
    outA[(size_t)(b*Sv + t)*Ev + i] = (__bf16)o;
  }
}

// ---------------------------------------------------------------------------
// Workspace layout (needs ~52.3 MB):
//   [0,16MB)   A bf16 (B*S x E)   -- reused as mixed-activation bf16 buffer
//   [16,18MB)  in_w  bf16
//   [18,20MB)  out_w bf16
//   [20,52MB)  Y fp32 (B*S x E)
//   [52MB,+256KB) chunk partial sums
extern "C" void kernel_launch(void* const* d_in, const int* in_sizes, int n_in,
                              void* d_out, int out_size, void* d_ws, size_t ws_size,
                              hipStream_t stream) {
  const float* x     = (const float*)d_in[0];
  const float* in_w  = (const float*)d_in[1];
  const float* in_b  = (const float*)d_in[2];
  const float* out_w = (const float*)d_in[3];
  const float* out_b = (const float*)d_in[4];
  const float* mix_w = (const float*)d_in[5];
  const float* mix_b = (const float*)d_in[6];
  float* out = (float*)d_out;

  char* ws = (char*)d_ws;
  __bf16* Abf   = (__bf16*)(ws);                        // 16 MB (also mixedA)
  __bf16* Winb  = (__bf16*)(ws + (size_t)(16u << 20));  // 2 MB
  __bf16* Woutb = (__bf16*)(ws + (size_t)(18u << 20));  // 2 MB
  float*  Y     = (float*) (ws + (size_t)(20u << 20));  // 32 MB
  float*  part  = (float*) (ws + (size_t)(52u << 20));  // 256 KB

  // 1) convert weights + transpose/convert activations to bf16
  k_conv_w<<<dim3((Ev*Ev + 255)/256), dim3(256), 0, stream>>>(in_w,  Winb,  Ev*Ev);
  k_conv_w<<<dim3((Ev*Ev + 255)/256), dim3(256), 0, stream>>>(out_w, Woutb, Ev*Ev);
  k_conv_x<<<dim3(Sv/32, Ev/32, Bv), dim3(32, 8), 0, stream>>>(x, Abf);

  // 2) in_proj GEMM (bf16 WMMA, fp32 accumulate): Y = A @ in_w^T + in_b
  k_gemm<0><<<dim3(Ev/BN, Mv/BM), dim3(256), 0, stream>>>(Abf, Winb, in_b, Y);

  // 3) causal head mixing as prefix sums (writes bf16 into Abf)
  k_mix_partial<<<dim3(Bv*Hv, NCHUNK), dim3(64), 0, stream>>>(Y, mix_w, part);
  k_mix_final  <<<dim3(Bv*Hv, NCHUNK), dim3(64), 0, stream>>>(Y, mix_w, mix_b, part, Abf);

  // 4) out_proj GEMM with fused (b t e)->(b e t) transpose in the epilogue
  k_gemm<1><<<dim3(Ev/BN, Mv/BM), dim3(256), 0, stream>>>(Abf, Woutb, out_b, out);
}
